// Level1AttentionLayer_71751723647565
// MI455X (gfx1250) — matrix operand order repaired and checked by reading, not compiled, not run on hardware
//
#include <hip/hip_runtime.h>

// ---------------------------------------------------------------------------
// Pairwise inverse-distance kernel for MI455X (gfx1250, wave32, WMMA).
// out[b,n,m] = 1 / (1 + sqrt(max(|x_n|^2 + |y_m|^2 - 2 x_n.y_m, 1e-7)))
// B=16, N=M=2048, D=128, fp32 in/out.
// GEMM core: bf16x3 split (hi*hi + hi*lo + lo*hi) on v_wmma_f32_16x16x32_bf16.
// Norms: re-read of WGP$-hot fp32 rows (1 row per thread), pure FMA.
// ---------------------------------------------------------------------------

typedef __attribute__((ext_vector_type(16))) __bf16 v16bf;
typedef __attribute__((ext_vector_type(8)))  __bf16 v8bf;
typedef __attribute__((ext_vector_type(8)))  float  v8f;

#define BATCH 16
#define NDIM  2048
#define KD    128
#define BT    128          // block tile (rows and cols)
#define LDSS  136          // padded LDS stride in bf16 elements (272B: conflict-free)
#define EPS   1e-7f

// A fragment: lane holds row (lane&15); halves 0..7 -> k0..k0+7, 8..15 -> k0+16..k0+23
// with k0 = 8*(lane>=16).  Two 16B LDS loads.
__device__ __forceinline__ v16bf load_fragA(const __bf16* base, int row, int khalf) {
  const __bf16* p = base + row * LDSS + khalf;
  v8bf c0 = *(const v8bf*)(p);
  v8bf c1 = *(const v8bf*)(p + 16);
  v16bf r;
#pragma unroll
  for (int i = 0; i < 8; ++i) { r[i] = c0[i]; r[i + 8] = c1[i]; }
  return r;
}

// B fragment: lane holds col (lane&15); halves 0..7 -> k0..k0+7, 8..15 -> k0+16..k0+23
// with k0 = 8*(lane>=16).  Two 16B LDS loads.
__device__ __forceinline__ v16bf load_fragB(const __bf16* base, int col, int khalf) {
  const __bf16* p = base + col * LDSS + khalf;
  v8bf c0 = *(const v8bf*)(p);
  v8bf c1 = *(const v8bf*)(p + 16);
  v16bf r;
#pragma unroll
  for (int i = 0; i < 8; ++i) { r[i] = c0[i]; r[i + 8] = c1[i]; }
  return r;
}

// Truncation split: hi = top16(f), lo = top16(f - hi).  lo absorbs the
// truncation error, so hi+lo carries ~16-17 mantissa bits.  Pure int ops;
// two halves pack into one dword ((u0>>16)|(u1&0xFFFF0000) == v_perm_b32).
__device__ __forceinline__ void split_store(__bf16* hiArr, __bf16* loArr,
                                            int r, int c4, float4 v) {
  const unsigned u0 = __float_as_uint(v.x);
  const unsigned u1 = __float_as_uint(v.y);
  const unsigned u2 = __float_as_uint(v.z);
  const unsigned u3 = __float_as_uint(v.w);
  uint2 hv;
  hv.x = (u0 >> 16) | (u1 & 0xFFFF0000u);
  hv.y = (u2 >> 16) | (u3 & 0xFFFF0000u);
  const float r0 = v.x - __uint_as_float(u0 & 0xFFFF0000u);
  const float r1 = v.y - __uint_as_float(u1 & 0xFFFF0000u);
  const float r2 = v.z - __uint_as_float(u2 & 0xFFFF0000u);
  const float r3 = v.w - __uint_as_float(u3 & 0xFFFF0000u);
  const unsigned l0 = __float_as_uint(r0);
  const unsigned l1 = __float_as_uint(r1);
  const unsigned l2 = __float_as_uint(r2);
  const unsigned l3 = __float_as_uint(r3);
  uint2 lv;
  lv.x = (l0 >> 16) | (l1 & 0xFFFF0000u);
  lv.y = (l2 >> 16) | (l3 & 0xFFFF0000u);
  *(uint2*)(hiArr + r * LDSS + c4 * 4) = hv;
  *(uint2*)(loArr + r * LDSS + c4 * 4) = lv;
}

__launch_bounds__(256)
__global__ void pairwise_invdist_kernel(const float* __restrict__ X,
                                        const float* __restrict__ Y,
                                        float* __restrict__ out) {
  __shared__ __bf16 sAhi[BT * LDSS];
  __shared__ __bf16 sAlo[BT * LDSS];
  __shared__ __bf16 sBhi[BT * LDSS];
  __shared__ __bf16 sBlo[BT * LDSS];
  __shared__ float  sXsq[BT];
  __shared__ float  sYsq[BT];

  const int tid     = threadIdx.x;
  const int b       = blockIdx.z;
  const int tileRow = blockIdx.y * BT;   // X rows
  const int tileCol = blockIdx.x * BT;   // Y rows (output cols)

  const float* Xt = X + ((size_t)b * NDIM + tileRow) * KD;
  const float* Yt = Y + ((size_t)b * NDIM + tileCol) * KD;

  // ---- Stage X/Y tiles into LDS as bf16 hi/lo splits ----------------------
  {
    const int col4 = tid & 31;           // float4 column (k = col4*4)
    const int rowg = tid >> 5;           // 0..7
#pragma unroll
    for (int i = 0; i < 16; ++i) {
      const int r = rowg + i * 8;
      float4 vx = *(const float4*)(Xt + (size_t)r * KD + col4 * 4);
      float4 vy = *(const float4*)(Yt + (size_t)r * KD + col4 * 4);
      split_store(sAhi, sAlo, r, col4, vx);
      split_store(sBhi, sBlo, r, col4, vy);
    }
  }

  // ---- Squared norms: 1 row per thread, fp32 re-read (lines are WGP$-hot) -
  {
    const int r = tid & 127;
    const float* src = ((tid < 128) ? Xt : Yt) + (size_t)r * KD;
    float s = 0.0f;
#pragma unroll
    for (int j = 0; j < 32; ++j) {
      float4 v = *(const float4*)(src + j * 4);
      s = fmaf(v.x, v.x, s);
      s = fmaf(v.y, v.y, s);
      s = fmaf(v.z, v.z, s);
      s = fmaf(v.w, v.w, s);
    }
    if (tid < 128) sXsq[r] = s; else sYsq[r] = s;
  }

  __syncthreads();   // tiles + norms visible to every wave

  // ---- WMMA main loop -----------------------------------------------------
  const int lane    = tid & 31;
  const int wave    = tid >> 5;
  const int rowBase = (wave >> 2) * 64;  // 2 wave-rows * 64
  const int colBase = (wave & 3) * 32;   // 4 wave-cols * 32
  const int fr      = lane & 15;
  const int kselA   = (lane >> 4) * 8;
  const int kselB   = (lane >> 4) * 8;

  v8f acc[4][2];
#pragma unroll
  for (int ti = 0; ti < 4; ++ti)
#pragma unroll
    for (int tj = 0; tj < 2; ++tj)
      acc[ti][tj] = (v8f){};

#pragma unroll
  for (int ks = 0; ks < 4; ++ks) {
    v16bf aHi[4], aLo[4], bHi[2], bLo[2];
#pragma unroll
    for (int ti = 0; ti < 4; ++ti) {
      const int r = rowBase + ti * 16 + fr;
      aHi[ti] = load_fragA(sAhi, r, ks * 32 + kselA);
      aLo[ti] = load_fragA(sAlo, r, ks * 32 + kselA);
    }
#pragma unroll
    for (int tj = 0; tj < 2; ++tj) {
      const int c = colBase + tj * 16 + fr;
      bHi[tj] = load_fragB(sBhi, c, ks * 32 + kselB);
      bLo[tj] = load_fragB(sBlo, c, ks * 32 + kselB);
    }
#pragma unroll
    for (int ti = 0; ti < 4; ++ti) {
#pragma unroll
      for (int tj = 0; tj < 2; ++tj) {
        acc[ti][tj] = __builtin_amdgcn_wmma_f32_16x16x32_bf16(
            false, aHi[ti], false, bHi[tj], (short)0, acc[ti][tj], false, false);
        acc[ti][tj] = __builtin_amdgcn_wmma_f32_16x16x32_bf16(
            false, aHi[ti], false, bLo[tj], (short)0, acc[ti][tj], false, false);
        acc[ti][tj] = __builtin_amdgcn_wmma_f32_16x16x32_bf16(
            false, aLo[ti], false, bHi[tj], (short)0, acc[ti][tj], false, false);
      }
    }
  }

  // ---- Epilogue: d2 = xsq + ysq - 2*dot; out = 1/(1+sqrt(d2)) -------------
  const int halfSel = (lane >> 4) * 8;
#pragma unroll
  for (int ti = 0; ti < 4; ++ti) {
#pragma unroll
    for (int tj = 0; tj < 2; ++tj) {
      const int cLocal = colBase + tj * 16 + fr;
      const float ysq  = sYsq[cLocal];
      const size_t gColBase = (size_t)tileCol + cLocal;
#pragma unroll
      for (int j = 0; j < 8; ++j) {
        const int rLocal = rowBase + ti * 16 + j + halfSel;
        const float xsq = sXsq[rLocal];
        float d2 = fmaxf(fmaf(-2.0f, acc[ti][tj][j], xsq + ysq), EPS);
        float s  = __builtin_amdgcn_sqrtf(d2);
        float o  = __builtin_amdgcn_rcpf(1.0f + s);
        out[((size_t)b * NDIM + tileRow + rLocal) * NDIM + gColBase] = o;
      }
    }
  }
}

extern "C" void kernel_launch(void* const* d_in, const int* in_sizes, int n_in,
                              void* d_out, int out_size, void* d_ws, size_t ws_size,
                              hipStream_t stream) {
  (void)in_sizes; (void)n_in; (void)out_size; (void)d_ws; (void)ws_size;
  const float* X = (const float*)d_in[0];
  const float* Y = (const float*)d_in[1];
  float* out     = (float*)d_out;
  dim3 grid(NDIM / BT, NDIM / BT, BATCH);   // 16 x 16 x 16
  dim3 block(256);
  pairwise_invdist_kernel<<<grid, block, 0, stream>>>(X, Y, out);
}
